// ParallelLayer_69028714381403
// MI455X (gfx1250) — matrix operand order repaired and checked by reading, not compiled, not run on hardware
//
#include <hip/hip_runtime.h>
#include <math.h>

// ---------------------------------------------------------------------------
// CDNA5 (gfx1250) implementation of the bipartite-GNN "ParallelLayer".
// GEMMs run as bf16->f32 WMMA (v_wmma_f32_16x16x32_bf16), BN folded into
// per-column affine applied during bf16 tile conversion. All on `stream`.
// ---------------------------------------------------------------------------

#define NHID 128
#define EPSV 1e-5f

typedef __attribute__((ext_vector_type(16))) __bf16 v16bf;
typedef __attribute__((ext_vector_type(8)))  float  v8f;

union FragAB { v16bf v; __bf16 h[16]; uint4 q[2]; };
union PkBf4  { __bf16 h[4]; uint2 u; };

// -------------------------- WMMA GEMM ---------------------------------------
// Y[M,N] = act(A[M,K]) @ W[K,N], act = optional per-K-column affine+relu.
// Block: 128 threads = 4 waves. Tile: 64(M) x 64(N), K-chunks of 32.
// Register-staged pipeline: global loads of chunk k+1 overlap WMMA on chunk k.
#define TM 64
#define TN 64
#define TK 32
#define LPAD 40   // bf16 elements per LDS row (32 data + 8 pad, 80B stride)

template<bool AFF>
__global__ __launch_bounds__(128)
void gemm_wmma_kernel(const float* __restrict__ A, const float* __restrict__ W,
                      float* __restrict__ Y, int M, int K, int N,
                      const float* __restrict__ aff /* scale[K],shift[K] */)
{
  __shared__ __align__(16) __bf16 sA[TM * LPAD];
  __shared__ __align__(16) __bf16 sB[TN * LPAD];
  const int tid  = threadIdx.x;
  const int lane = tid & 31;
  const int wav  = tid >> 5;
  const int hi   = lane >> 4;
  const int lo   = lane & 15;
  const int mBase = blockIdx.x * TM;
  const int nBase = blockIdx.y * TN;

  // A staging: fixed 4-col group acol, rows am0+16j. Invalid rows read row 0
  // (harmless: WMMA row m only feeds D row m, and invalid rows are not stored).
  const int acol = (tid & 7) * 4;
  const int am0  = tid >> 3;
  const float* arow[4];
  #pragma unroll
  for (int j = 0; j < 4; ++j) {
    int gm = mBase + am0 + 16 * j;
    arow[j] = A + (size_t)(gm < M ? gm : 0) * K + acol;
  }
  // W staging: fixed 4-wide n group wn, k rows wk0+8j.
  const int wn  = (tid & 15) * 4;
  const int wk0 = tid >> 4;

  float4 ra[4], rw[4];

  auto loadChunk = [&](int k0) {
    float4 sc, sh;
    if (AFF) {
      sc = *(const float4*)(aff + k0 + acol);
      sh = *(const float4*)(aff + K + k0 + acol);
    }
    #pragma unroll
    for (int j = 0; j < 4; ++j) {
      float4 v = *(const float4*)(arow[j] + k0);
      if (AFF) {
        v.x = fmaxf(v.x * sc.x + sh.x, 0.f);
        v.y = fmaxf(v.y * sc.y + sh.y, 0.f);
        v.z = fmaxf(v.z * sc.z + sh.z, 0.f);
        v.w = fmaxf(v.w * sc.w + sh.w, 0.f);
      }
      ra[j] = v;
    }
    #pragma unroll
    for (int j = 0; j < 4; ++j)
      rw[j] = *(const float4*)(W + (size_t)(k0 + wk0 + 8 * j) * N + nBase + wn);
  };
  auto storeChunk = [&]() {
    #pragma unroll
    for (int j = 0; j < 4; ++j) {
      PkBf4 p;
      p.h[0] = (__bf16)ra[j].x; p.h[1] = (__bf16)ra[j].y;
      p.h[2] = (__bf16)ra[j].z; p.h[3] = (__bf16)ra[j].w;
      *(uint2*)&sA[(am0 + 16 * j) * LPAD + acol] = p.u;
    }
    #pragma unroll
    for (int j = 0; j < 4; ++j) {
      int kk = wk0 + 8 * j;
      sB[(wn + 0) * LPAD + kk] = (__bf16)rw[j].x;
      sB[(wn + 1) * LPAD + kk] = (__bf16)rw[j].y;
      sB[(wn + 2) * LPAD + kk] = (__bf16)rw[j].z;
      sB[(wn + 3) * LPAD + kk] = (__bf16)rw[j].w;
    }
  };

  v8f acc[4];
  #pragma unroll
  for (int t = 0; t < 4; ++t)
    #pragma unroll
    for (int r = 0; r < 8; ++r) acc[t][r] = 0.f;

  loadChunk(0);
  for (int k0 = 0; k0 < K; k0 += TK) {
    storeChunk();
    __syncthreads();
    if (k0 + TK < K) loadChunk(k0 + TK);   // overlap next-chunk loads with WMMA

    // A fragment: lane (hi,lo): row lo,
    // h[j] = K hi*16 + j
    FragAB fa;
    {
      const __bf16* row = &sA[(wav * 16 + lo) * LPAD];
      fa.q[0] = *(const uint4*)&row[hi * 16];
      fa.q[1] = *(const uint4*)&row[hi * 16 + 8];
    }
    #pragma unroll
    for (int t = 0; t < 4; ++t) {
      // B fragment: lane (hi,lo): col lo, h[j] = K hi*16 + j
      FragAB fb;
      const __bf16* row = &sB[(t * 16 + lo) * LPAD];
      fb.q[0] = *(const uint4*)&row[hi * 16];
      fb.q[1] = *(const uint4*)&row[hi * 16 + 8];
      acc[t] = __builtin_amdgcn_wmma_f32_16x16x32_bf16(false, fa.v, false, fb.v,
                                                       (short)0, acc[t], false, false);
    }
    __syncthreads();
  }

  // C/D layout: VGPR r -> row hi*8+r, lanes -> col lo
  #pragma unroll
  for (int t = 0; t < 4; ++t)
    #pragma unroll
    for (int r = 0; r < 8; ++r) {
      int gm = mBase + wav * 16 + hi * 8 + r;
      if (gm < M) Y[(size_t)gm * N + (nBase + t * 16 + lo)] = acc[t][r];
    }
}

// GEMM over the virtual concatenation cat = [x | right_full | left_full | rl_full]
// (K=256, N=256), each 64-wide piece with its own BN affine+relu.
// TK=32 divides the 64-wide pieces, so piece selection is per-chunk uniform.
// M = 2n is a multiple of 64 -> no M guards.
__global__ __launch_bounds__(128)
void gemm_cat_wmma_kernel(const float* __restrict__ x64, const float* __restrict__ ri64,
                          const float* __restrict__ li64, const float* __restrict__ rli64,
                          const float* __restrict__ a5, const float* __restrict__ a6,
                          const float* __restrict__ a7, const float* __restrict__ a8,
                          const float* __restrict__ W, float* __restrict__ Y,
                          int nHalf, int M)
{
  const int K = 256, N = 256;
  __shared__ __align__(16) __bf16 sA[TM * LPAD];
  __shared__ __align__(16) __bf16 sB[TN * LPAD];
  const int tid  = threadIdx.x;
  const int lane = tid & 31;
  const int wav  = tid >> 5;
  const int hi   = lane >> 4;
  const int lo   = lane & 15;
  const int mBase = blockIdx.x * TM;
  const int nBase = blockIdx.y * TN;

  const int acol = (tid & 7) * 4;
  const int am0  = tid >> 3;
  const float* rowX[4]; const float* row1[4]; const float* row2[4]; const float* row3[4];
  const float* af1[4];  const float* af2[4];  const float* af3[4];
  #pragma unroll
  for (int j = 0; j < 4; ++j) {
    int gm = mBase + am0 + 16 * j;
    bool left = gm < nHalf;
    rowX[j] = x64 + (size_t)gm * 64;
    row1[j] = left ? ri64 + (size_t)gm * 64 : rowX[j];
    af1[j]  = left ? a6 : a5;
    row2[j] = left ? rowX[j] : li64 + (size_t)(gm - nHalf) * 64;
    af2[j]  = left ? a5 : a7;
    row3[j] = left ? rowX[j] : rli64 + (size_t)(gm - nHalf) * 64;
    af3[j]  = left ? a5 : a8;
  }
  const int wn  = (tid & 15) * 4;
  const int wk0 = tid >> 4;

  float4 ra[4], rw[4];

  auto loadChunk = [&](int k0) {
    int p  = k0 >> 6;
    int kc = (k0 & 63) + acol;
    #pragma unroll
    for (int j = 0; j < 4; ++j) {
      const float* base = (p == 0) ? rowX[j] : (p == 1) ? row1[j]
                        : (p == 2) ? row2[j] : row3[j];
      const float* ab   = (p == 0) ? a5 : (p == 1) ? af1[j]
                        : (p == 2) ? af2[j] : af3[j];
      float4 v  = *(const float4*)(base + kc);
      float4 sc = *(const float4*)(ab + kc);
      float4 sh = *(const float4*)(ab + 64 + kc);
      v.x = fmaxf(v.x * sc.x + sh.x, 0.f);
      v.y = fmaxf(v.y * sc.y + sh.y, 0.f);
      v.z = fmaxf(v.z * sc.z + sh.z, 0.f);
      v.w = fmaxf(v.w * sc.w + sh.w, 0.f);
      ra[j] = v;
    }
    #pragma unroll
    for (int j = 0; j < 4; ++j)
      rw[j] = *(const float4*)(W + (size_t)(k0 + wk0 + 8 * j) * N + nBase + wn);
  };
  auto storeChunk = [&]() {
    #pragma unroll
    for (int j = 0; j < 4; ++j) {
      PkBf4 p;
      p.h[0] = (__bf16)ra[j].x; p.h[1] = (__bf16)ra[j].y;
      p.h[2] = (__bf16)ra[j].z; p.h[3] = (__bf16)ra[j].w;
      *(uint2*)&sA[(am0 + 16 * j) * LPAD + acol] = p.u;
    }
    #pragma unroll
    for (int j = 0; j < 4; ++j) {
      int kk = wk0 + 8 * j;
      sB[(wn + 0) * LPAD + kk] = (__bf16)rw[j].x;
      sB[(wn + 1) * LPAD + kk] = (__bf16)rw[j].y;
      sB[(wn + 2) * LPAD + kk] = (__bf16)rw[j].z;
      sB[(wn + 3) * LPAD + kk] = (__bf16)rw[j].w;
    }
  };

  v8f acc[4];
  #pragma unroll
  for (int t = 0; t < 4; ++t)
    #pragma unroll
    for (int r = 0; r < 8; ++r) acc[t][r] = 0.f;

  loadChunk(0);
  for (int k0 = 0; k0 < K; k0 += TK) {
    storeChunk();
    __syncthreads();
    if (k0 + TK < K) loadChunk(k0 + TK);

    FragAB fa;
    {
      const __bf16* row = &sA[(wav * 16 + lo) * LPAD];
      fa.q[0] = *(const uint4*)&row[hi * 16];
      fa.q[1] = *(const uint4*)&row[hi * 16 + 8];
    }
    #pragma unroll
    for (int t = 0; t < 4; ++t) {
      FragAB fb;
      const __bf16* row = &sB[(t * 16 + lo) * LPAD];
      fb.q[0] = *(const uint4*)&row[hi * 16];
      fb.q[1] = *(const uint4*)&row[hi * 16 + 8];
      acc[t] = __builtin_amdgcn_wmma_f32_16x16x32_bf16(false, fa.v, false, fb.v,
                                                       (short)0, acc[t], false, false);
    }
    __syncthreads();
  }
  #pragma unroll
  for (int t = 0; t < 4; ++t)
    #pragma unroll
    for (int r = 0; r < 8; ++r) {
      int gm = mBase + wav * 16 + hi * 8 + r;
      Y[(size_t)gm * N + (nBase + t * 16 + lo)] = acc[t][r];
    }
}

// --------------------- BatchNorm column statistics --------------------------
__global__ void colstats_kernel(const float* __restrict__ Y, int M, int N,
                                int rowsPerBlock, float* __restrict__ st)
{
  int r0 = blockIdx.x * rowsPerBlock;
  int r1 = min(r0 + rowsPerBlock, M);
  for (int c = threadIdx.x; c < N; c += blockDim.x) {
    float s = 0.f, sq = 0.f;
    for (int r = r0; r < r1; ++r) {
      float v = Y[(size_t)r * N + c];
      s += v; sq += v * v;
    }
    atomicAdd(&st[c], s);
    atomicAdd(&st[N + c], sq);
  }
}

__global__ void finalize_affine_kernel(const float* __restrict__ st,
                                       const float* __restrict__ gamma,
                                       const float* __restrict__ beta,
                                       int N, float Minv, float* __restrict__ aff)
{
  int c = blockIdx.x * blockDim.x + threadIdx.x;
  if (c >= N) return;
  float mu  = st[c] * Minv;
  float var = fmaxf(st[N + c] * Minv - mu * mu, 0.f);
  float rs  = rsqrtf(var + EPSV);
  float g = gamma ? gamma[c] : 1.f;
  float b = beta  ? beta[c]  : 0.f;
  float scale = g * rs;
  aff[c]     = scale;
  aff[N + c] = b - mu * scale;
}

// --------------------- Edge gather / segment scatter ------------------------
__device__ __forceinline__ void atomicMaxFloat(float* addr, float v) {
  if (v >= 0.f) atomicMax((int*)addr, __float_as_int(v));
  else          atomicMin((unsigned int*)addr, __float_as_uint(v));
}

// msg = act(X[ridx[e]+rowOff]) (128 feats); out[oidx[e]] gets sum(c) / max(c+256).
// 4 threads per edge (32 feats each).
__global__ void scatter_kernel(const float* __restrict__ X,
                               const int* __restrict__ ridx, int rowOff,
                               const int* __restrict__ oidx,
                               const float* __restrict__ aff,
                               float* __restrict__ out, float* __restrict__ cnt,
                               int nE)
{
  long t = blockIdx.x * (long)blockDim.x + threadIdx.x;
  long e = t >> 2;
  if (e >= nE) return;
  int g = (int)(t & 3);
  int r = ridx[e] + rowOff;
  size_t o = (size_t)oidx[e] * 384;
  const float4* x4 = (const float4*)(X + (size_t)r * 128 + g * 32);
  float* osum = out + o + g * 32;
  float* omax = osum + 256;
  #pragma unroll
  for (int q = 0; q < 8; ++q) {
    float4 v = x4[q];
    float vv[4] = {v.x, v.y, v.z, v.w};
    #pragma unroll
    for (int j = 0; j < 4; ++j) {
      float f = vv[j];
      if (aff) {
        int cc = g * 32 + q * 4 + j;
        f = fmaxf(f * aff[cc] + aff[128 + cc], 0.f);
      }
      atomicAdd(&osum[q * 4 + j], f);
      atomicMaxFloat(&omax[q * 4 + j], f);
    }
  }
  if (g == 0 && cnt) atomicAdd(&cnt[oidx[e]], 1.f);
}

// mean = sum / max(cnt,1); max <- 0 where cnt == 0.
__global__ void gfin_kernel(float* __restrict__ out, const float* __restrict__ cnt, int n)
{
  long t = blockIdx.x * (long)blockDim.x + threadIdx.x;
  if (t >= (long)n * 128) return;
  int i = (int)(t >> 7), c = (int)(t & 127);
  float ct = cnt[i];
  float* row = out + (size_t)i * 384;
  row[128 + c] = row[c] / fmaxf(ct, 1.f);
  if (!(ct > 0.f)) row[256 + c] = 0.f;
}

// --------------------- Init / epilogue --------------------------------------
__global__ void init_gscatter_kernel(float* __restrict__ p, size_t count)
{ // rows of 384: cols [0,256)=0, [256,384)=-inf (for atomic max)
  size_t i = blockIdx.x * (size_t)blockDim.x + threadIdx.x;
  if (i >= count) return;
  int c = (int)(i % 384);
  p[i] = (c < 256) ? 0.f : __int_as_float(0xff800000u);
}

__global__ void zero_kernel(float* __restrict__ p, size_t count)
{
  size_t i = blockIdx.x * (size_t)blockDim.x + threadIdx.x;
  if (i < count) p[i] = 0.f;
}

__global__ void epilogue_kernel(const float* __restrict__ Yr, const float* __restrict__ aff,
                                int M, int N, float* __restrict__ out)
{
  size_t i = blockIdx.x * (size_t)blockDim.x + threadIdx.x;
  if (i >= (size_t)M * N) return;
  int c = (int)(i % N);
  out[i] = fmaxf(Yr[i] * aff[c] + aff[N + c], 0.f);
}

// ---------------------------------------------------------------------------
extern "C" void kernel_launch(void* const* d_in, const int* in_sizes, int n_in,
                              void* d_out, int out_size, void* d_ws, size_t ws_size,
                              hipStream_t stream) {
  (void)n_in; (void)out_size; (void)ws_size;
  const float* xs  = (const float*)d_in[0];
  const int*   bip = (const int*)d_in[2];
  const float* W1a = (const float*)d_in[3];
  const float* g1a = (const float*)d_in[4];
  const float* b1a = (const float*)d_in[5];
  const float* W1b = (const float*)d_in[6];
  const float* g1b = (const float*)d_in[7];
  const float* b1b = (const float*)d_in[8];
  const float* W2a = (const float*)d_in[9];
  const float* g2a = (const float*)d_in[10];
  const float* b2a = (const float*)d_in[11];
  const float* W2b = (const float*)d_in[12];
  const float* g2b = (const float*)d_in[13];
  const float* b2b = (const float*)d_in[14];
  const float* Wl1 = (const float*)d_in[15];
  const float* Wl2 = (const float*)d_in[17];
  const float* Wl3 = (const float*)d_in[19];
  const float* Wl4 = (const float*)d_in[21];
  const float* Woa = (const float*)d_in[23];
  const float* goa = (const float*)d_in[24];
  const float* boa = (const float*)d_in[25];
  const float* Wob = (const float*)d_in[26];
  const float* gob = (const float*)d_in[27];
  const float* bob = (const float*)d_in[28];
  // Note: bl1..bl4 biases cancel inside BatchNorm (constant per-column shift).

  const int nTotal = in_sizes[0] / NHID;   // 2n
  const int n      = nTotal / 2;
  const int E      = in_sizes[2] / 2;
  const int* srcI  = bip;
  const int* dstI  = bip + E;

  // ---------------- workspace layout (f32) ----------------
  float* w = (float*)d_ws;
  size_t off = 0;
  auto alloc = [&](size_t c) { float* p = w + off; off += c; return p; };
  float* ri    = alloc((size_t)n * 384);     // right_info       (also reused by oa)
  float* li    = alloc((size_t)n * 384);     // left_info
  float* rli   = alloc((size_t)n * 384);     // right_left_info  (also reused by ob)
  float* h1    = alloc((size_t)nTotal * 128);
  float* xl    = alloc((size_t)nTotal * 128);
  float* h2a   = alloc((size_t)n * 384);
  float* rin   = alloc((size_t)n * 128);
  float* x64   = alloc((size_t)nTotal * 64);
  float* ri64  = alloc((size_t)n * 64);
  float* li64  = alloc((size_t)n * 64);
  float* rli64 = alloc((size_t)n * 64);
  float* cntS  = alloc((size_t)n);
  float* cntD  = alloc((size_t)n);
  float* stats = alloc(10 * 768);
  float* affsB = alloc(10 * 768);
  float* oa = ri;   // 2n*256 over dead ri(+li) region
  float* ob = rli;  // 2n*128 over dead rli region

  auto st  = [&](int s) { return stats + (size_t)s * 768; };
  auto aff = [&](int s) { return affsB + (size_t)s * 768; };

  auto runGemm = [&](const float* A, const float* Wm, float* Y,
                     int M, int K, int N, const float* af) {
    dim3 g((M + TM - 1) / TM, N / TN);
    if (af) gemm_wmma_kernel<true ><<<g, 128, 0, stream>>>(A, Wm, Y, M, K, N, af);
    else    gemm_wmma_kernel<false><<<g, 128, 0, stream>>>(A, Wm, Y, M, K, N, nullptr);
  };
  auto runStats = [&](const float* Y, int M, int N,
                      const float* ga, const float* be, int s) {
    const int rpb = 1024;
    colstats_kernel<<<(M + rpb - 1) / rpb, 128, 0, stream>>>(Y, M, N, rpb, st(s));
    finalize_affine_kernel<<<(N + 127) / 128, 128, 0, stream>>>(st(s), ga, be, N,
                                                                1.f / (float)M, aff(s));
  };
  auto runScatter = [&](const float* X, const int* rIdx, int rOff, const int* oIdx,
                        const float* af, float* out, float* cnt) {
    long t = 4L * E;
    scatter_kernel<<<(unsigned)((t + 255) / 256), 256, 0, stream>>>(X, rIdx, rOff, oIdx,
                                                                    af, out, cnt, E);
  };
  auto runGfin = [&](float* out, const float* cnt) {
    long t = (long)n * 128;
    gfin_kernel<<<(unsigned)((t + 255) / 256), 256, 0, stream>>>(out, cnt, n);
  };

  // ---------------- init -------------------------------------------------
  {
    size_t c = (size_t)3 * n * 384;   // ri,li,rli contiguous
    init_gscatter_kernel<<<(unsigned)((c + 255) / 256), 256, 0, stream>>>(ri, c);
    size_t z = (size_t)2 * n + 10 * 768;  // cntS,cntD,stats contiguous
    zero_kernel<<<(unsigned)((z + 255) / 256), 256, 0, stream>>>(cntS, z);
  }

  // pass 1: right_info = gscatter(xs[n+dst], src)
  runScatter(xs, dstI, n, srcI, nullptr, ri, cntS);

  // MLP1 on combined_xs (stats over all 2n rows)
  runGemm(xs, W1a, h1, nTotal, 128, 128, nullptr);
  runStats(h1, nTotal, 128, g1a, b1a, 0);
  runGemm(h1, W1b, xl, nTotal, 128, 128, aff(0));
  runStats(xl, nTotal, 128, g1b, b1b, 1);

  // pass 2: left_info = gscatter(relu(bn(xl))[src], dst)
  runScatter(xl, srcI, 0, dstI, aff(1), li, cntD);

  // finalize right_info (mean/max) before it is consumed
  runGfin(ri, cntS);

  // MLP2 on right_info
  runGemm(ri, W2a, h2a, n, 384, 384, nullptr);
  runStats(h2a, n, 384, g2a, b2a, 2);
  runGemm(h2a, W2b, rin, n, 384, 128, aff(2));
  runStats(rin, n, 128, g2b, b2b, 3);

  // pass 3: right_left_info = gscatter(relu(bn(rin))[src], dst)  (same dst counts)
  runScatter(rin, srcI, 0, dstI, aff(3), rli, nullptr);
  runGfin(li, cntD);
  runGfin(rli, cntD);

  // four 64-wide projections (biases cancel in BN)
  runGemm(xs,  Wl1, x64,   nTotal, 128, 64, nullptr); runStats(x64,   nTotal, 64, nullptr, nullptr, 4);
  runGemm(ri,  Wl2, ri64,  n,      384, 64, nullptr); runStats(ri64,  n,      64, nullptr, nullptr, 5);
  runGemm(li,  Wl3, li64,  n,      384, 64, nullptr); runStats(li64,  n,      64, nullptr, nullptr, 6);
  runGemm(rli, Wl4, rli64, n,      384, 64, nullptr); runStats(rli64, n,      64, nullptr, nullptr, 7);

  // output MLP over virtual concat (K=256 -> N=256 -> N=128)
  {
    dim3 g(nTotal / TM, 4);
    gemm_cat_wmma_kernel<<<g, 128, 0, stream>>>(x64, ri64, li64, rli64,
                                                aff(4), aff(5), aff(6), aff(7),
                                                Woa, oa, n, nTotal);
  }
  runStats(oa, nTotal, 256, goa, boa, 8);
  runGemm(oa, Wob, ob, nTotal, 256, 128, aff(8));
  runStats(ob, nTotal, 128, gob, bob, 9);

  {
    size_t t = (size_t)nTotal * 128;
    epilogue_kernel<<<(unsigned)((t + 255) / 256), 256, 0, stream>>>(ob, aff(9),
                                                                     nTotal, 128,
                                                                     (float*)d_out);
  }
}
